// Time_LSTM_1_91027536871510
// MI455X (gfx1250) — matrix operand order repaired and checked by reading, not compiled, not run on hardware
//
#include <hip/hip_runtime.h>

#define BDIM 16384
#define DIN  256
#define HDIM 512
#define TPAD 32

typedef __attribute__((ext_vector_type(16))) __bf16 v16bf;
typedef __attribute__((ext_vector_type(8)))  __bf16 v8bf;
typedef __attribute__((ext_vector_type(8)))  float  v8f;

static __device__ __forceinline__ v8f wmma_bf16(v16bf a, v16bf b, v8f c) {
    return __builtin_amdgcn_wmma_f32_16x16x32_bf16(
        /*neg_a=*/false, a, /*neg_b=*/false, b,
        /*c_mod=*/(short)0, c, /*reuse_a=*/false, /*reuse_b=*/false);
}

// A-matrix (16x32 bf16, row-major):
// lane L (L<16): M=L, K = 0..15; lane L+16: M=L, K = 16..31
static __device__ __forceinline__ v16bf load_A(const __bf16* __restrict__ row, int k0, int hi) {
    return *reinterpret_cast<const v16bf*>(row + k0 + 16 * hi);
}

// B-matrix (32x16 bf16): lane L (L<16): N=L, K=k0..k0+15; lane L+16: N=L, K=k0+16..k0+31
// W row-major (out_dim, K): B[k][n] = W[n][k] -> 16 contiguous K per lane.
static __device__ __forceinline__ v16bf load_B(const __bf16* __restrict__ w, int stride,
                                               int nrow, int k0, int hi) {
    return *reinterpret_cast<const v16bf*>(w + (size_t)nrow * stride + k0 + 16 * hi);
}

static __device__ __forceinline__ float sigf(float x) { return 1.0f / (1.0f + __expf(-x)); }

__global__ void cvt_f32_bf16(const float* __restrict__ s, __bf16* __restrict__ d, int n) {
    int i = blockIdx.x * blockDim.x + threadIdx.x;
    if (i < n) d[i] = (__bf16)s[i];
}

// copy rows of width src_cols into rows of width dst_cols, zero-padding the tail
__global__ void cvt_pad_rows(const float* __restrict__ s, __bf16* __restrict__ d,
                             int rows, int src_cols, int dst_cols) {
    int i = blockIdx.x * blockDim.x + threadIdx.x;
    int n = rows * dst_cols;
    if (i < n) {
        int r = i / dst_cols, c = i - r * dst_cols;
        d[i] = (c < src_cols) ? (__bf16)s[r * src_cols + c] : (__bf16)0.0f;
    }
}

// Epilogue for one 16x16 C/D tile: VGPR j, lanes 0-15 -> M=j, N=lane;
// lanes 16-31 -> M=j+8, N=lane-16.
static __device__ __forceinline__ void epilogue_tile(
    v8f ai, v8f af, v8f at, v8f att, v8f ak, v8f ao,
    int row0, int col, int hi,
    float bi, float bf, float bt, float bc, float bo,
    const float* __restrict__ c_prev32, float* __restrict__ out)
{
    const size_t BH = (size_t)BDIM * HDIM;
    #pragma unroll
    for (int j = 0; j < 8; ++j) {
        const int row = row0 + j + 8 * hi;
        const size_t idx = (size_t)row * HDIM + col;
        float iv = sigf(ai[j] + bi);
        float fv = sigf(af[j] + bf);
        float Tv = sigf(at[j] + bt + sigf(att[j]));
        float kv = tanhf(ak[j] + bc);
        float cn = iv * Tv * kv + fv * c_prev32[idx];
        float ov = sigf(ao[j] + bo);
        out[idx]          = ov * tanhf(cn);   // h_next
        out[BH + idx]     = cn;               // c_next
        out[2 * BH + idx] = Tv;               // T
    }
}

__global__ void __launch_bounds__(256)
time_lstm_wmma(const __bf16* __restrict__ xb, const __bf16* __restrict__ hb,
               const __bf16* __restrict__ cb, const __bf16* __restrict__ db,
               const __bf16* __restrict__ Wix, const __bf16* __restrict__ Wfx,
               const __bf16* __restrict__ Wtx, const __bf16* __restrict__ Wcx,
               const __bf16* __restrict__ Wox,
               const __bf16* __restrict__ Wih, const __bf16* __restrict__ Wfh,
               const __bf16* __restrict__ Wch, const __bf16* __restrict__ Woh,
               const __bf16* __restrict__ Wic, const __bf16* __restrict__ Wfc,
               const __bf16* __restrict__ Woc,
               const __bf16* __restrict__ Wtt, const __bf16* __restrict__ Wot,
               const float* __restrict__ bix, const float* __restrict__ bfx,
               const float* __restrict__ btx, const float* __restrict__ bcx,
               const float* __restrict__ box,
               const float* __restrict__ c_prev32, float* __restrict__ out)
{
    const int wave = threadIdx.x >> 5;
    const int lane = threadIdx.x & 31;
    // All 8 waves of a block share ONE weight column tile (B operand identical ->
    // WGP$/L2 hits) and span 8 * 32 = 256 distinct rows. Consecutive blocks walk
    // the row dimension with the same column tile (64 blocks/column), keeping the
    // ~157 KB weight slice hot in L2 across the whole row sweep.
    const int colT = blockIdx.x >> 6;              // 32 column tiles
    const int rowG = blockIdx.x & 63;              // 64 row groups of 256 rows
    const int row0 = rowG * 256 + wave * 32;       // this wave: rows [row0, row0+32)
    const int col0 = colT * 16;
    const int n  = lane & 15;                      // A row / B column within tile
    const int hi = lane >> 4;

    // Two row tiles per wave: every loaded B fragment feeds 2 WMMAs.
    v8f ai0 = {}, ai1 = {}, af0 = {}, af1 = {}, at0 = {}, at1 = {};
    v8f att0 = {}, att1 = {}, ak0 = {}, ak1 = {}, ao0 = {}, ao1 = {};

    // ---- Phase 1: x projections, K = 256 ----
    const __bf16* xrow0 = xb + (size_t)(row0 + n) * DIN;
    const __bf16* xrow1 = xrow0 + (size_t)16 * DIN;
    #pragma unroll
    for (int k0 = 0; k0 < DIN; k0 += 32) {
        v16bf a0 = load_A(xrow0, k0, hi);
        v16bf a1 = load_A(xrow1, k0, hi);
        v16bf b;
        b = load_B(Wix, DIN, col0 + n, k0, hi); ai0 = wmma_bf16(a0, b, ai0); ai1 = wmma_bf16(a1, b, ai1);
        b = load_B(Wfx, DIN, col0 + n, k0, hi); af0 = wmma_bf16(a0, b, af0); af1 = wmma_bf16(a1, b, af1);
        b = load_B(Wtx, DIN, col0 + n, k0, hi); at0 = wmma_bf16(a0, b, at0); at1 = wmma_bf16(a1, b, at1);
        b = load_B(Wcx, DIN, col0 + n, k0, hi); ak0 = wmma_bf16(a0, b, ak0); ak1 = wmma_bf16(a1, b, ak1);
        b = load_B(Wox, DIN, col0 + n, k0, hi); ao0 = wmma_bf16(a0, b, ao0); ao1 = wmma_bf16(a1, b, ao1);
    }

    // ---- Phase 2: h and c_prev projections, K = 512 ----
    const __bf16* hrow0 = hb + (size_t)(row0 + n) * HDIM;
    const __bf16* hrow1 = hrow0 + (size_t)16 * HDIM;
    const __bf16* crow0 = cb + (size_t)(row0 + n) * HDIM;
    const __bf16* crow1 = crow0 + (size_t)16 * HDIM;
    #pragma unroll 2
    for (int k0 = 0; k0 < HDIM; k0 += 32) {
        v16bf ah0 = load_A(hrow0, k0, hi);
        v16bf ah1 = load_A(hrow1, k0, hi);
        v16bf ac0 = load_A(crow0, k0, hi);
        v16bf ac1 = load_A(crow1, k0, hi);
        v16bf b;
        b = load_B(Wih, HDIM, col0 + n, k0, hi); ai0 = wmma_bf16(ah0, b, ai0); ai1 = wmma_bf16(ah1, b, ai1);
        b = load_B(Wfh, HDIM, col0 + n, k0, hi); af0 = wmma_bf16(ah0, b, af0); af1 = wmma_bf16(ah1, b, af1);
        b = load_B(Wch, HDIM, col0 + n, k0, hi); ak0 = wmma_bf16(ah0, b, ak0); ak1 = wmma_bf16(ah1, b, ak1);
        b = load_B(Woh, HDIM, col0 + n, k0, hi); ao0 = wmma_bf16(ah0, b, ao0); ao1 = wmma_bf16(ah1, b, ao1);
        b = load_B(Wic, HDIM, col0 + n, k0, hi); ai0 = wmma_bf16(ac0, b, ai0); ai1 = wmma_bf16(ac1, b, ai1);
        b = load_B(Wfc, HDIM, col0 + n, k0, hi); af0 = wmma_bf16(ac0, b, af0); af1 = wmma_bf16(ac1, b, af1);
        b = load_B(Woc, HDIM, col0 + n, k0, hi); ao0 = wmma_bf16(ac0, b, ao0); ao1 = wmma_bf16(ac1, b, ao1);
    }

    // ---- Phase 3: delt projections, K = 16 zero-padded to 32 ----
    {
        const __bf16* drow0 = db + (size_t)(row0 + n) * TPAD;
        const __bf16* drow1 = drow0 + (size_t)16 * TPAD;
        v16bf ad0 = load_A(drow0, 0, hi);
        v16bf ad1 = load_A(drow1, 0, hi);
        v16bf b;
        b = load_B(Wtt, TPAD, col0 + n, 0, hi); att0 = wmma_bf16(ad0, b, att0); att1 = wmma_bf16(ad1, b, att1);
        b = load_B(Wot, TPAD, col0 + n, 0, hi); ao0 = wmma_bf16(ad0, b, ao0); ao1 = wmma_bf16(ad1, b, ao1);
    }

    // ---- Epilogue: gates + cell update in fp32 ----
    const int col = col0 + n;
    const float bi = bix[col], bf = bfx[col], bt = btx[col], bc = bcx[col], bo = box[col];
    epilogue_tile(ai0, af0, at0, att0, ak0, ao0, row0,      col, hi, bi, bf, bt, bc, bo, c_prev32, out);
    epilogue_tile(ai1, af1, at1, att1, ak1, ao1, row0 + 16, col, hi, bi, bf, bt, bc, bo, c_prev32, out);
}

extern "C" void kernel_launch(void* const* d_in, const int* in_sizes, int n_in,
                              void* d_out, int out_size, void* d_ws, size_t ws_size,
                              hipStream_t stream) {
    const float* x      = (const float*)d_in[0];
    const float* h      = (const float*)d_in[1];
    const float* c_prev = (const float*)d_in[2];
    const float* delt   = (const float*)d_in[3];
    const float* W_ix = (const float*)d_in[4];  const float* b_ix = (const float*)d_in[5];
    const float* W_ih = (const float*)d_in[6];  const float* W_ic = (const float*)d_in[7];
    const float* W_fx = (const float*)d_in[8];  const float* b_fx = (const float*)d_in[9];
    const float* W_fh = (const float*)d_in[10]; const float* W_fc = (const float*)d_in[11];
    const float* W_tx = (const float*)d_in[12]; const float* b_tx = (const float*)d_in[13];
    const float* W_tt = (const float*)d_in[14];
    const float* W_cx = (const float*)d_in[15]; const float* b_cx = (const float*)d_in[16];
    const float* W_ch = (const float*)d_in[17];
    const float* W_ox = (const float*)d_in[18]; const float* b_ox = (const float*)d_in[19];
    const float* W_oh = (const float*)d_in[20]; const float* W_oc = (const float*)d_in[21];
    const float* W_ot = (const float*)d_in[22];

    char* ws = (char*)d_ws;
    size_t off = 0;
    auto carve = [&](size_t elems) -> __bf16* {
        __bf16* p = (__bf16*)(ws + off);
        off = (off + elems * sizeof(__bf16) + 255) & ~(size_t)255;
        return p;
    };

    __bf16* xb  = carve((size_t)BDIM * DIN);
    __bf16* hb  = carve((size_t)BDIM * HDIM);
    __bf16* cb  = carve((size_t)BDIM * HDIM);
    __bf16* db  = carve((size_t)BDIM * TPAD);
    __bf16* Wix = carve((size_t)HDIM * DIN);
    __bf16* Wfx = carve((size_t)HDIM * DIN);
    __bf16* Wtx = carve((size_t)HDIM * DIN);
    __bf16* Wcx = carve((size_t)HDIM * DIN);
    __bf16* Wox = carve((size_t)HDIM * DIN);
    __bf16* Wih = carve((size_t)HDIM * HDIM);
    __bf16* Wfh = carve((size_t)HDIM * HDIM);
    __bf16* Wch = carve((size_t)HDIM * HDIM);
    __bf16* Woh = carve((size_t)HDIM * HDIM);
    __bf16* Wic = carve((size_t)HDIM * HDIM);
    __bf16* Wfc = carve((size_t)HDIM * HDIM);
    __bf16* Woc = carve((size_t)HDIM * HDIM);
    __bf16* Wtt = carve((size_t)HDIM * TPAD);
    __bf16* Wot = carve((size_t)HDIM * TPAD);

    auto cvt = [&](const float* s, __bf16* d, int nElem) {
        cvt_f32_bf16<<<(nElem + 255) / 256, 256, 0, stream>>>(s, d, nElem);
    };
    cvt(x, xb, BDIM * DIN);
    cvt(h, hb, BDIM * HDIM);
    cvt(c_prev, cb, BDIM * HDIM);
    cvt_pad_rows<<<(BDIM * TPAD + 255) / 256, 256, 0, stream>>>(delt, db, BDIM, 16, TPAD);
    cvt(W_ix, Wix, HDIM * DIN);
    cvt(W_fx, Wfx, HDIM * DIN);
    cvt(W_tx, Wtx, HDIM * DIN);
    cvt(W_cx, Wcx, HDIM * DIN);
    cvt(W_ox, Wox, HDIM * DIN);
    cvt(W_ih, Wih, HDIM * HDIM);
    cvt(W_fh, Wfh, HDIM * HDIM);
    cvt(W_ch, Wch, HDIM * HDIM);
    cvt(W_oh, Woh, HDIM * HDIM);
    cvt(W_ic, Wic, HDIM * HDIM);
    cvt(W_fc, Wfc, HDIM * HDIM);
    cvt(W_oc, Woc, HDIM * HDIM);
    cvt_pad_rows<<<(HDIM * TPAD + 255) / 256, 256, 0, stream>>>(W_tt, Wtt, HDIM, 16, TPAD);
    cvt_pad_rows<<<(HDIM * TPAD + 255) / 256, 256, 0, stream>>>(W_ot, Wot, HDIM, 16, TPAD);

    // Grid: 32 column tiles * 64 row groups (256 rows each) = 2048 blocks.
    // blockIdx low bits = row group -> consecutive blocks share the weight slice.
    const int blocks = (HDIM / 16) * (BDIM / 256);
    time_lstm_wmma<<<blocks, 256, 0, stream>>>(
        xb, hb, cb, db,
        Wix, Wfx, Wtx, Wcx, Wox,
        Wih, Wfh, Wch, Woh,
        Wic, Wfc, Woc,
        Wtt, Wot,
        b_ix, b_fx, b_tx, b_cx, b_ox,
        c_prev, (float*)d_out);
}
